// SingleQuantumCircuit_33595234189578
// MI455X (gfx1250) — matrix operand order repaired and checked by reading, not compiled, not run on hardware
//
#include <hip/hip_runtime.h>

// ---------------------------------------------------------------------------
// MI455X (gfx1250) implementation of the hybrid classical/quantum circuit.
//   k_convert -> k_gemm1 (WMMA f16) -> k_gemm2 (WMMA f16) -> k_expm -> k_sim
// State vector (128 KB / batch element) lives entirely in LDS (320 KB/WGP).
// ---------------------------------------------------------------------------

typedef __attribute__((ext_vector_type(16))) _Float16 v16h;
typedef __attribute__((ext_vector_type(8)))  float    v8f;

#define N_QUBITS 14
#define BATCH    128
#define NSTATE   (1 << N_QUBITS)     // 16384 complex amplitudes
#define NGEN     15
#define NGROUPS  13
#define NOBS     13
#define VDEPTH   4
#define ENC_DIM  (NGROUPS * NGEN)    // 195
#define HID      256
#define NPAD     208                 // 195 padded to 13 tiles of 16

__device__ __forceinline__ float2 cmulf(float2 a, float2 b) {
  return make_float2(a.x * b.x - a.y * b.y, a.x * b.y + a.y * b.x);
}

// ------------------------------- convert -----------------------------------
__global__ __launch_bounds__(256) void k_convert(
    const float* __restrict__ x, const float* __restrict__ W1,
    const float* __restrict__ W2, _Float16* __restrict__ XH,
    _Float16* __restrict__ W1H, _Float16* __restrict__ W2H) {
  int i = blockIdx.x * 256 + threadIdx.x;          // 0..65535
  if (i < BATCH * 256) XH[i] = (_Float16)x[i];
  W1H[i] = (_Float16)W1[i];                        // 256*256
  if (i < NPAD * 256) {
    int n = i >> 8, k = i & 255;
    W2H[i] = (n < ENC_DIM) ? (_Float16)W2[n * 256 + k] : (_Float16)0.0f;
  }
}

// --------------------------- WMMA GEMM helpers -----------------------------
// A tile (16x32 f16): lane lo=row M; halves[0..15]=K(16*hi .. 16*hi+15)
// B tile (32x16 f16): lane lo=col N; halves[0..15]=K(16*hi .. 16*hi+15)
// C/D (v8f): lane lo=col N; c[i] = row (i + 8*hi)
__device__ __forceinline__ v8f wmma_tile_k256(const _Float16* __restrict__ arow,
                                              const _Float16* __restrict__ brow,
                                              int hi) {
  v8f c = {};
  for (int k0 = 0; k0 < 256; k0 += 32) {
    v16h a, b;
#pragma unroll
    for (int j = 0; j < 16; ++j) a[j] = arow[k0 + 16 * hi + j];
#pragma unroll
    for (int j = 0; j < 16; ++j) b[j] = brow[k0 + 16 * hi + j];
    c = __builtin_amdgcn_wmma_f32_16x16x32_f16(false, a, false, b,
                                               (short)0, c, false, false);
  }
  return c;
}

// GEMM1: H1 = silu(x @ W1^T + b1), M=128 N=256 K=256.  8x16 tiles, 1/wave.
__global__ __launch_bounds__(256) void k_gemm1(
    const _Float16* __restrict__ XH, const _Float16* __restrict__ W1H,
    const float* __restrict__ b1, _Float16* __restrict__ H1H) {
  int wave = threadIdx.x >> 5, lane = threadIdx.x & 31;
  int lo = lane & 15, hi = lane >> 4;
  int tile = blockIdx.x * 8 + wave;                // 0..127
  int tM = tile & 7, tN = tile >> 3;
  v8f c = wmma_tile_k256(XH + (tM * 16 + lo) * 256,
                         W1H + (tN * 16 + lo) * 256, hi);
  int n = tN * 16 + lo;
  float bias = b1[n];
#pragma unroll
  for (int i = 0; i < 8; ++i) {
    int m = tM * 16 + i + 8 * hi;
    float v = c[i] + bias;
    float sg = v / (1.0f + __expf(-v));            // SiLU
    H1H[m * 256 + n] = (_Float16)sg;
  }
}

// GEMM2: enc = H1 @ W2^T + b2, M=128 N=195(->208) K=256.
__global__ __launch_bounds__(256) void k_gemm2(
    const _Float16* __restrict__ H1H, const _Float16* __restrict__ W2H,
    const float* __restrict__ b2, float* __restrict__ ENC) {
  int wave = threadIdx.x >> 5, lane = threadIdx.x & 31;
  int lo = lane & 15, hi = lane >> 4;
  int tile = blockIdx.x * 8 + wave;                // 0..103
  int tM = tile & 7, tN = tile >> 3;               // tN in 0..12
  v8f c = wmma_tile_k256(H1H + (tM * 16 + lo) * 256,
                         W2H + (tN * 16 + lo) * 256, hi);
  int n = tN * 16 + lo;
  if (n < ENC_DIM) {
    float bias = b2[n];
#pragma unroll
    for (int i = 0; i < 8; ++i) {
      int m = tM * 16 + i + 8 * hi;
      ENC[m * ENC_DIM + n] = c[i] + bias;
    }
  }
}

// --------------------- SU(4) unitaries: U = exp(i*A) -----------------------
__device__ __constant__ float PAULI_R[4][2][2] = {
    {{1.f, 0.f}, {0.f, 1.f}},    // I
    {{0.f, 1.f}, {1.f, 0.f}},    // X
    {{0.f, 0.f}, {0.f, 0.f}},    // Y (real part)
    {{1.f, 0.f}, {0.f, -1.f}}};  // Z
__device__ __constant__ float PAULI_I[4][2][2] = {
    {{0.f, 0.f}, {0.f, 0.f}},
    {{0.f, 0.f}, {0.f, 0.f}},
    {{0.f, -1.f}, {1.f, 0.f}},   // Y (imag part)
    {{0.f, 0.f}, {0.f, 0.f}}};

__device__ __forceinline__ void mm4(const float Ar[4][4], const float Ai[4][4],
                                    const float Br[4][4], const float Bi[4][4],
                                    float Cr[4][4], float Ci[4][4]) {
#pragma unroll
  for (int i = 0; i < 4; ++i)
#pragma unroll
    for (int j = 0; j < 4; ++j) {
      float sr = 0.f, si = 0.f;
#pragma unroll
      for (int k = 0; k < 4; ++k) {
        sr += Ar[i][k] * Br[k][j] - Ai[i][k] * Bi[k][j];
        si += Ar[i][k] * Bi[k][j] + Ai[i][k] * Br[k][j];
      }
      Cr[i][j] = sr;
      Ci[i][j] = si;
    }
}

__global__ __launch_bounds__(128) void k_expm(const float* __restrict__ enc,
                                              float2* __restrict__ UM) {
  int tid = blockIdx.x * blockDim.x + threadIdx.x;
  if (tid >= BATCH * NGROUPS) return;
  int b = tid / NGROUPS, g = tid % NGROUPS;
  const float* th = enc + b * ENC_DIM + g * NGEN;

  // A = sum_g theta_g * kron(P[a], P[b]) (Hermitian)
  float Ar[4][4] = {{0}}, Ai[4][4] = {{0}};
#pragma unroll
  for (int ge = 0; ge < NGEN; ++ge) {
    int pa = (ge + 1) >> 2, pb = (ge + 1) & 3;
    float t = th[ge];
#pragma unroll
    for (int i = 0; i < 4; ++i)
#pragma unroll
      for (int j = 0; j < 4; ++j) {
        float ar = PAULI_R[pa][i >> 1][j >> 1], ai = PAULI_I[pa][i >> 1][j >> 1];
        float br = PAULI_R[pb][i & 1][j & 1],   bi = PAULI_I[pb][i & 1][j & 1];
        Ar[i][j] += t * (ar * br - ai * bi);
        Ai[i][j] += t * (ar * bi + ai * br);
      }
  }
  // M = i*A (anti-Hermitian), exp via scaling-and-squaring + Taylor.
  float Mr[4][4], Mi[4][4];
#pragma unroll
  for (int i = 0; i < 4; ++i)
#pragma unroll
    for (int j = 0; j < 4; ++j) { Mr[i][j] = -Ai[i][j]; Mi[i][j] = Ar[i][j]; }

  float nrm = 0.f;
#pragma unroll
  for (int i = 0; i < 4; ++i) {
    float rs = 0.f;
#pragma unroll
    for (int j = 0; j < 4; ++j) rs += fabsf(Mr[i][j]) + fabsf(Mi[i][j]);
    nrm = fmaxf(nrm, rs);
  }
  int s = 0;
  while (nrm > 0.5f && s < 24) { nrm *= 0.5f; ++s; }
  float sc = exp2f(-(float)s);
#pragma unroll
  for (int i = 0; i < 4; ++i)
#pragma unroll
    for (int j = 0; j < 4; ++j) { Mr[i][j] *= sc; Mi[i][j] *= sc; }

  float Ur[4][4], Ui[4][4], Tr[4][4], Ti[4][4];
#pragma unroll
  for (int i = 0; i < 4; ++i)
#pragma unroll
    for (int j = 0; j < 4; ++j) {
      float id = (i == j) ? 1.f : 0.f;
      Ur[i][j] = id; Ui[i][j] = 0.f; Tr[i][j] = id; Ti[i][j] = 0.f;
    }
  for (int k = 1; k <= 12; ++k) {
    float Nr[4][4], Ni[4][4];
    mm4(Tr, Ti, Mr, Mi, Nr, Ni);
    float inv = 1.f / (float)k;
#pragma unroll
    for (int i = 0; i < 4; ++i)
#pragma unroll
      for (int j = 0; j < 4; ++j) {
        Tr[i][j] = Nr[i][j] * inv;  Ti[i][j] = Ni[i][j] * inv;
        Ur[i][j] += Tr[i][j];       Ui[i][j] += Ti[i][j];
      }
  }
  for (int it = 0; it < s; ++it) {
    float Nr[4][4], Ni[4][4];
    mm4(Ur, Ui, Ur, Ui, Nr, Ni);
#pragma unroll
    for (int i = 0; i < 4; ++i)
#pragma unroll
      for (int j = 0; j < 4; ++j) { Ur[i][j] = Nr[i][j]; Ui[i][j] = Ni[i][j]; }
  }
  float2* o = UM + tid * 16;
#pragma unroll
  for (int i = 0; i < 4; ++i)
#pragma unroll
    for (int j = 0; j < 4; ++j) o[i * 4 + j] = make_float2(Ur[i][j], Ui[i][j]);
}

// ---------------------- state-vector simulation (LDS) ----------------------
__global__ __launch_bounds__(1024) void k_sim(
    const float2* __restrict__ UM, const float* __restrict__ vp,
    const float* __restrict__ oA, const float* __restrict__ oB,
    const float* __restrict__ oD, float* __restrict__ out) {
  extern __shared__ float2 smem[];
  float2* st  = smem;                          // 16384 amplitudes (128 KB)
  float2* Us  = smem + NSTATE;                 // 16 (current SU4 gate)
  float2* Hs  = smem + NSTATE + 16;            // 13*16 observable matrices
  float*  red = (float*)(smem + NSTATE + 16 + NOBS * 16);  // 1024 floats

  const int b = blockIdx.x;
  const int tid = threadIdx.x;

  for (int i = tid; i < NSTATE; i += 1024) st[i] = make_float2(0.f, 0.f);
  if (tid == 0) st[0] = make_float2(1.f, 0.f);

  // Build Hermitian observables: H = h + h^dagger (lower tri = A+iB, diag = D)
  if (tid < NOBS * 16) {
    int i = tid >> 4, e = tid & 15, r = e >> 2, cc = e & 3;
    float2 h;
    if (r == cc) {
      h = make_float2((r < 3) ? 2.f * oD[i * 4 + r + 1] : 0.f, 0.f);
    } else if (r > cc) {
      int idx = (r * (r - 1)) / 2 + cc;
      h = make_float2(oA[i * 6 + idx], oB[i * 6 + idx]);
    } else {
      int idx = (cc * (cc - 1)) / 2 + r;
      h = make_float2(oA[i * 6 + idx], -oB[i * 6 + idx]);
    }
    Hs[tid] = h;
  }
  __syncthreads();

  // ---- SU(4) brick-layer encoding gates (per-batch unitaries) ----
  const int SUNQ[NGROUPS] = {0, 2, 4, 6, 8, 10, 12, 1, 3, 5, 7, 9, 11};
#pragma unroll 1
  for (int g = 0; g < NGROUPS; ++g) {
    if (tid < 16) Us[tid] = UM[(b * NGROUPS + g) * 16 + tid];
    __syncthreads();
    int sh = 12 - SUNQ[g];
    int rest = 1 << sh;
    for (int t = tid; t < 4096; t += 1024) {
      int p = t >> sh, r = t & (rest - 1);
      int base = (p << (sh + 2)) + r;
      float2 v[4];
#pragma unroll
      for (int k = 0; k < 4; ++k) v[k] = st[base + k * rest];
#pragma unroll
      for (int k = 0; k < 4; ++k) {
        float2 acc = make_float2(0.f, 0.f);
#pragma unroll
        for (int l = 0; l < 4; ++l) {
          float2 m = cmulf(Us[k * 4 + l], v[l]);
          acc.x += m.x; acc.y += m.y;
        }
        st[base + k * rest] = acc;
      }
    }
    __syncthreads();
  }

  // ---- variational ansatz: RY layers + CNOT chains ----
#pragma unroll 1
  for (int d = 0; d < VDEPTH; ++d) {
#pragma unroll 1
    for (int q = 0; q < N_QUBITS; ++q) {
      float half = 0.5f * vp[d * N_QUBITS + q];
      float cs = cosf(half), sn = sinf(half);
      int sh = 13 - q;
      int rest = 1 << sh;
      for (int t = tid; t < 8192; t += 1024) {
        int p = t >> sh, r = t & (rest - 1);
        int i0 = (p << (sh + 1)) + r, i1 = i0 + rest;
        float2 a0 = st[i0], a1 = st[i1];
        st[i0] = make_float2(cs * a0.x - sn * a1.x, cs * a0.y - sn * a1.y);
        st[i1] = make_float2(sn * a0.x + cs * a1.x, sn * a0.y + cs * a1.y);
      }
      __syncthreads();
    }
#pragma unroll 1
    for (int q = 0; q < N_QUBITS - 1; ++q) {     // CNOT(q -> q+1): swap k=2,3
      int sh = 12 - q;
      int rest = 1 << sh;
      for (int t = tid; t < 4096; t += 1024) {
        int p = t >> sh, r = t & (rest - 1);
        int base = (p << (sh + 2)) + r;
        float2 x2 = st[base + 2 * rest], x3 = st[base + 3 * rest];
        st[base + 2 * rest] = x3;
        st[base + 3 * rest] = x2;
      }
      __syncthreads();
    }
  }

  // ---- expectation values of 2-local Hermitians on sliding windows ----
#pragma unroll 1
  for (int i = 0; i < NOBS; ++i) {
    int sh = 12 - i;
    int rest = 1 << sh;
    float partial = 0.f;
    for (int t = tid; t < 4096; t += 1024) {
      int p = t >> sh, r = t & (rest - 1);
      int base = (p << (sh + 2)) + r;
      float2 v[4];
#pragma unroll
      for (int k = 0; k < 4; ++k) v[k] = st[base + k * rest];
#pragma unroll
      for (int k = 0; k < 4; ++k) {
        float2 w = make_float2(0.f, 0.f);
#pragma unroll
        for (int l = 0; l < 4; ++l) {
          float2 m = cmulf(Hs[i * 16 + k * 4 + l], v[l]);
          w.x += m.x; w.y += m.y;
        }
        partial += v[k].x * w.x + v[k].y * w.y;  // Re(conj(v_k) * w)
      }
    }
    red[tid] = partial;
    __syncthreads();
    for (int off = 512; off > 0; off >>= 1) {
      if (tid < off) red[tid] += red[tid + off];
      __syncthreads();
    }
    if (tid == 0) out[b * NOBS + i] = red[0];
    __syncthreads();
  }
}

// ------------------------------- launcher ----------------------------------
extern "C" void kernel_launch(void* const* d_in, const int* in_sizes, int n_in,
                              void* d_out, int out_size, void* d_ws,
                              size_t ws_size, hipStream_t stream) {
  const float* x  = (const float*)d_in[0];
  const float* W1 = (const float*)d_in[1];
  const float* b1 = (const float*)d_in[2];
  const float* W2 = (const float*)d_in[3];
  const float* b2 = (const float*)d_in[4];
  const float* vp = (const float*)d_in[5];
  const float* oA = (const float*)d_in[6];
  const float* oB = (const float*)d_in[7];
  const float* oD = (const float*)d_in[8];
  float* out = (float*)d_out;

  // Workspace layout (bytes), total ~682 KB.
  char* ws = (char*)d_ws;
  _Float16* XH  = (_Float16*)(ws + 0);        //  65536 B : x as f16
  _Float16* W1H = (_Float16*)(ws + 65536);    // 131072 B : W1 as f16
  _Float16* W2H = (_Float16*)(ws + 196608);   // 106496 B : W2 f16, padded 208 rows
  _Float16* H1H = (_Float16*)(ws + 303104);   //  65536 B : hidden f16
  float*    ENC = (float*)   (ws + 368640);   //  99840 B : enc f32 (128x195)
  float2*   UM  = (float2*)  (ws + 468736);   // 212992 B : 1664 SU(4) unitaries

  k_convert<<<256, 256, 0, stream>>>(x, W1, W2, XH, W1H, W2H);
  k_gemm1<<<16, 256, 0, stream>>>(XH, W1H, b1, H1H);
  k_gemm2<<<13, 256, 0, stream>>>(H1H, W2H, b2, ENC);
  k_expm<<<13, 128, 0, stream>>>(ENC, UM);

  size_t smem = (size_t)(NSTATE + 16 + NOBS * 16) * sizeof(float2) +
                1024 * sizeof(float);          // ~134 KB of 320 KB/WGP
  k_sim<<<BATCH, 1024, smem, stream>>>(UM, vp, oA, oB, oD, out);
}
